// GraphFormer_1864015806553
// MI455X (gfx1250) — matrix operand order repaired and checked by reading, not compiled, not run on hardware
//
#include <hip/hip_runtime.h>
#include <hip/hip_bf16.h>

typedef __attribute__((ext_vector_type(16))) _Float16 v16h;
typedef __attribute__((ext_vector_type(8)))  _Float16 v8h;
typedef __attribute__((ext_vector_type(8)))  float    v8f;
typedef unsigned int u32x4 __attribute__((ext_vector_type(4)));
typedef int          i32x4 __attribute__((ext_vector_type(4)));
typedef int          i32x8 __attribute__((ext_vector_type(8)));

#define R_    3
#define N0_   80000
#define N1_   20000
#define B_    4096
#define E1_   300000
#define E2_   65536
#define FIN_  128
#define HC1_  256
#define HC2_  128

static inline int cdiv(int a, int b){ return (a + b - 1) / b; }

// ---- monotone float<->uint encoding for atomicMax-based segment max ----
__device__ __forceinline__ unsigned fenc(float f){
  unsigned u = __float_as_uint(f);
  return (u & 0x80000000u) ? ~u : (u | 0x80000000u);
}
__device__ __forceinline__ float fdec(unsigned e){
  unsigned b = (e & 0x80000000u) ? (e & 0x7FFFFFFFu) : ~e;
  return __uint_as_float(b);
}

// ---------------- fill ----------------
__global__ void k_fill(unsigned* __restrict__ p, unsigned v, int n){
  int i = blockIdx.x * blockDim.x + threadIdx.x;
  int stride = gridDim.x * blockDim.x;
  for (; i < n; i += stride) p[i] = v;
}

// ---------------- gather features -> f16 ----------------
__global__ void k_gather(const float* __restrict__ feat, const int* __restrict__ ids,
                         _Float16* __restrict__ X, int n0){
  int idx = blockIdx.x * blockDim.x + threadIdx.x;   // over n0*128
  if (idx >= n0 * FIN_) return;
  int node = idx >> 7, c = idx & 127;
  int id = ids[node];
  X[idx] = (_Float16)feat[(size_t)id * FIN_ + c];
}

// ---------------- cast + transpose weights: W[K,N] f32 -> Wt[N,K] f16 ----------------
__global__ void k_castT(const float* __restrict__ W, _Float16* __restrict__ Wt, int K, int N){
  int idx = blockIdx.x * blockDim.x + threadIdx.x;
  if (idx >= K * N) return;
  int k = idx / N, n = idx % N;
  Wt[(size_t)n * K + k] = (_Float16)W[idx];
}

// ---------------- WMMA GEMM: C[M,N] = A[M,K](f16) @ Wt[N,K]^T(f16) + bias ----------------
// Whole weight matrix Wt[N,K] staged to LDS via TDM (tensor_load_to_lds) once per block.
// Each wave computes a 32x64 block of C: 2 M-tiles x 4 N-tiles, 8 WMMAs per 32-wide K step.
__global__ void k_gemm_wmma(const _Float16* __restrict__ A, const _Float16* __restrict__ Bt,
                            const float* __restrict__ bias, float* __restrict__ C,
                            int M, int N, int K){
  extern __shared__ _Float16 bsh[];   // N*K halves (<= 64KB)
  const int tid  = threadIdx.x;
  const int lane = tid & 31;
  const int wid  = tid >> 5;

#if __has_builtin(__builtin_amdgcn_tensor_load_to_lds) && __has_builtin(__builtin_amdgcn_s_wait_tensorcnt)
  if (wid == 0){                      // wave-uniform; TDM ignores EXEC
    unsigned lds_base = (unsigned)(unsigned long long)(void*)&bsh[0];
    unsigned long long ga = (unsigned long long)(const void*)Bt;
    u32x4 g0;
    g0.x = 1u;                                              // count=1 (valid), user mode
    g0.y = lds_base;                                        // lds_addr [63:32]
    g0.z = (unsigned)(ga & 0xFFFFFFFFu);                    // global_addr [95:64]
    g0.w = (unsigned)((ga >> 32) & 0x01FFFFFFu) | (2u << 30); // global_addr msbs | type=2
    i32x8 g1;
    g1[0] = 0x10000;                    // data_size=1 (2 bytes), no multicast/pad/iterate
    g1[1] = (int)((unsigned)K << 16);   // tensor_dim0[15:0] in [31:16]
    g1[2] = (int)((unsigned)N << 16);   // tensor_dim1[15:0] in [31:16]
    g1[3] = (int)((unsigned)K << 16);   // tile_dim0 in [31:16]
    g1[4] = N;                          // tile_dim1
    g1[5] = K;                          // tensor_dim0_stride low 32
    g1[6] = 0; g1[7] = 0;
    i32x4 gz = {0, 0, 0, 0};
#if __has_include(<hip/amd_detail/amd_gfx1250_TDM.h>)
    i32x8 gz8 = {0, 0, 0, 0, 0, 0, 0, 0};
    __builtin_amdgcn_tensor_load_to_lds(g0, g1, gz, gz, gz8, 0);
#else
    __builtin_amdgcn_tensor_load_to_lds(g0, g1, gz, gz, 0);
#endif
    __builtin_amdgcn_s_wait_tensorcnt(0);
  }
#else
  for (int i = tid; i < (N * K) / 8; i += blockDim.x)
    ((v8h*)bsh)[i] = ((const v8h*)Bt)[i];
#endif
  __syncthreads();

  const int nGroups = N >> 6;          // 64-col groups
  const int mStrips = M >> 5;          // 32-row strips
  const int nb = blockIdx.x % nGroups;
  const int mb = blockIdx.x / nGroups;
  const int strip = mb * 8 + wid;      // wave-uniform guard below
  if (strip < mStrips){
    const int hi  = lane >> 4;         // lane group 0/1
    const int l15 = lane & 15;
    const int aoff = hi ? 16 : 0;      // A: lanes>=16 hold K {16..31}
    const int boff = hi ? 16 : 0;      // B: lanes>=16 hold K {16..31}
    const _Float16* arow0 = A + (size_t)((strip << 5) + l15) * K;
    const _Float16* arow1 = arow0 + (size_t)16 * K;
    const _Float16* bcol  = bsh + (size_t)((nb << 6) + l15) * K;
    v8f acc[2][4];
    #pragma unroll
    for (int mi = 0; mi < 2; ++mi)
      #pragma unroll
      for (int ni = 0; ni < 4; ++ni) acc[mi][ni] = (v8f){0,0,0,0,0,0,0,0};

    for (int kk = 0; kk < K; kk += 32){
      if (kk + 32 < K){
        __builtin_prefetch(arow0 + kk + 32, 0, 0);   // global_prefetch_b8
        __builtin_prefetch(arow1 + kk + 32, 0, 0);
      }
      v8h a00 = *(const v8h*)(arow0 + kk + aoff);
      v8h a01 = *(const v8h*)(arow0 + kk + 8 + aoff);
      v16h a0 = __builtin_shufflevector(a00, a01, 0,1,2,3,4,5,6,7,8,9,10,11,12,13,14,15);
      v8h a10 = *(const v8h*)(arow1 + kk + aoff);
      v8h a11 = *(const v8h*)(arow1 + kk + 8 + aoff);
      v16h a1 = __builtin_shufflevector(a10, a11, 0,1,2,3,4,5,6,7,8,9,10,11,12,13,14,15);
      #pragma unroll
      for (int ni = 0; ni < 4; ++ni){
        v16h b = *(const v16h*)(bcol + (size_t)(ni << 4) * K + kk + boff);
        acc[0][ni] = __builtin_amdgcn_wmma_f32_16x16x32_f16(false, a0, false, b, (short)0,
                                                            acc[0][ni], false, false);
        acc[1][ni] = __builtin_amdgcn_wmma_f32_16x16x32_f16(false, a1, false, b, (short)0,
                                                            acc[1][ni], false, false);
      }
    }
    const int rb = (strip << 5) + (hi << 3);   // C: lanes>=16 hold rows +8
    #pragma unroll
    for (int mi = 0; mi < 2; ++mi){
      #pragma unroll
      for (int ni = 0; ni < 4; ++ni){
        const int col = (nb << 6) + (ni << 4) + l15;
        const float bv = bias[col];
        #pragma unroll
        for (int g = 0; g < 8; ++g)
          C[(size_t)(rb + mi * 16 + g) * N + col] = acc[mi][ni][g] + bv;
      }
    }
  }
}

// ---------------- per-edge attention logits: wave per (edge, head) ----------------
__global__ void k_edge_alpha(const float* __restrict__ q, const float* __restrict__ kmat,
                             const int* __restrict__ src, const int* __restrict__ dst,
                             float* __restrict__ alpha, int E, int H){
  const int lane = threadIdx.x & 31;
  const int task = blockIdx.x * (blockDim.x >> 5) + (threadIdx.x >> 5);
  if (task >= E * H) return;
  const int e = task / H, h = task % H;
  const int HC = H * 128;
  const float4 qv = *(const float4*)(q    + (size_t)dst[e] * HC + h * 128 + lane * 4);
  const float4 kv = *(const float4*)(kmat + (size_t)src[e] * HC + h * 128 + lane * 4);
  float d = qv.x * kv.x + qv.y * kv.y + qv.z * kv.z + qv.w * kv.w;
  #pragma unroll
  for (int o = 16; o; o >>= 1) d += __shfl_xor(d, o, 32);
  if (lane == 0) alpha[task] = d * 0.08838834764831845f;   // 1/sqrt(128)
}

__global__ void k_edge_max(const float* __restrict__ alpha, const int* __restrict__ dst,
                           unsigned* __restrict__ amax_enc, int E, int H){
  int t = blockIdx.x * blockDim.x + threadIdx.x;
  if (t >= E * H) return;
  atomicMax(&amax_enc[dst[t / H] * H + (t % H)], fenc(alpha[t]));
}

__global__ void k_amax_dec(const unsigned* __restrict__ enc, float* __restrict__ amax, int n){
  int i = blockIdx.x * blockDim.x + threadIdx.x;
  if (i >= n) return;
  float f = fdec(enc[i]);
  amax[i] = isfinite(f) ? f : 0.0f;      // reference: non-finite segment max -> 0
}

__global__ void k_edge_exp(float* __restrict__ alpha, const int* __restrict__ dst,
                           const float* __restrict__ amax, float* __restrict__ denom,
                           int E, int H){
  int t = blockIdx.x * blockDim.x + threadIdx.x;
  if (t >= E * H) return;
  int di = dst[t / H] * H + (t % H);
  float p = __expf(alpha[t] - amax[di]);
  alpha[t] = p;
  atomicAdd(&denom[di], p);
}

// ---------------- weighted V scatter: wave per (edge, head), float4 per lane ----------------
__global__ void k_edge_accum(const float* __restrict__ p, const int* __restrict__ src,
                             const int* __restrict__ dst, const float* __restrict__ denom,
                             const float* __restrict__ v, float* __restrict__ out,
                             int E, int H){
  const int lane = threadIdx.x & 31;
  const int task = blockIdx.x * (blockDim.x >> 5) + (threadIdx.x >> 5);
  if (task >= E * H) return;
  const int e = task / H, h = task % H;
  const int HC = H * 128;
  const float w = p[task] / denom[dst[e] * H + h];
  const float4 vv = *(const float4*)(v + (size_t)src[e] * HC + h * 128 + lane * 4);
  float* op = out + (size_t)dst[e] * HC + h * 128 + lane * 4;
  atomicAdd(op + 0, w * vv.x);
  atomicAdd(op + 1, w * vv.y);
  atomicAdd(op + 2, w * vv.z);
  atomicAdd(op + 3, w * vv.w);
}

// ---------------- layer-1 tail: beta gate + BN(eval) + ELU -> f16 activations ----------------
__global__ void k_beta_bn_elu(const float* __restrict__ outm, const float* __restrict__ xr,
                              const float* __restrict__ wbeta,
                              const float* __restrict__ bng, const float* __restrict__ bnb,
                              const float* __restrict__ bnm, const float* __restrict__ bnv,
                              _Float16* __restrict__ h16, int Nt, int HC){
  const int lane = threadIdx.x & 31;
  const int t = blockIdx.x * (blockDim.x >> 5) + (threadIdx.x >> 5);
  if (t >= Nt) return;
  const float* orow = outm + (size_t)t * HC;
  const float* rrow = xr   + (size_t)t * HC;
  float acc = 0.f;
  for (int j = lane; j < HC; j += 32){
    float o = orow[j], r = rrow[j];
    acc += o * wbeta[j] + r * wbeta[HC + j] + (o - r) * wbeta[2 * HC + j];
  }
  #pragma unroll
  for (int o = 16; o; o >>= 1) acc += __shfl_xor(acc, o, 32);
  const float beta = 1.f / (1.f + __expf(-acc));
  for (int j = lane; j < HC; j += 32){
    float o = orow[j], r = rrow[j];
    float val = beta * r + (1.f - beta) * o;
    val = (val - bnm[j]) * rsqrtf(bnv[j] + 1e-5f) * bng[j] + bnb[j];
    val = val > 0.f ? val : (__expf(val) - 1.f);
    h16[(size_t)t * HC + j] = (_Float16)val;
  }
}

// ---------------- layer-2 tail: beta gate -> strided f32 output ----------------
__global__ void k_beta_out(const float* __restrict__ outm, const float* __restrict__ xr,
                           const float* __restrict__ wbeta,
                           float* __restrict__ dout, int ostride, int Nt, int HC){
  const int lane = threadIdx.x & 31;
  const int t = blockIdx.x * (blockDim.x >> 5) + (threadIdx.x >> 5);
  if (t >= Nt) return;
  const float* orow = outm + (size_t)t * HC;
  const float* rrow = xr   + (size_t)t * HC;
  float acc = 0.f;
  for (int j = lane; j < HC; j += 32){
    float o = orow[j], r = rrow[j];
    acc += o * wbeta[j] + r * wbeta[HC + j] + (o - r) * wbeta[2 * HC + j];
  }
  #pragma unroll
  for (int o = 16; o; o >>= 1) acc += __shfl_xor(acc, o, 32);
  const float beta = 1.f / (1.f + __expf(-acc));
  for (int j = lane; j < HC; j += 32){
    float o = orow[j], r = rrow[j];
    dout[(size_t)t * ostride + j] = beta * r + (1.f - beta) * o;
  }
}

extern "C" void kernel_launch(void* const* d_in, const int* in_sizes, int n_in,
                              void* d_out, int out_size, void* d_ws, size_t ws_size,
                              hipStream_t stream){
  (void)in_sizes; (void)n_in; (void)out_size; (void)ws_size;
  const float* features = (const float*)d_in[0];
  const int*   n_ids    = (const int*)d_in[1];
  const int*   ei1      = (const int*)d_in[2];
  const int*   ei2      = (const int*)d_in[3];
  const float* wq1 = (const float*)d_in[4];  const float* bq1 = (const float*)d_in[5];
  const float* wk1 = (const float*)d_in[6];  const float* bk1 = (const float*)d_in[7];
  const float* wv1 = (const float*)d_in[8];  const float* bv1 = (const float*)d_in[9];
  const float* ws1 = (const float*)d_in[10]; const float* bs1 = (const float*)d_in[11];
  const float* wbeta1 = (const float*)d_in[12];
  const float* bng = (const float*)d_in[13]; const float* bnb = (const float*)d_in[14];
  const float* bnm = (const float*)d_in[15]; const float* bnv = (const float*)d_in[16];
  const float* wq2 = (const float*)d_in[17]; const float* bq2 = (const float*)d_in[18];
  const float* wk2 = (const float*)d_in[19]; const float* bk2 = (const float*)d_in[20];
  const float* wv2 = (const float*)d_in[21]; const float* bv2 = (const float*)d_in[22];
  const float* ws2 = (const float*)d_in[23]; const float* bs2 = (const float*)d_in[24];
  const float* wbeta2 = (const float*)d_in[25];
  float* dout = (float*)d_out;

  // ---- workspace sub-allocation (256B aligned) ----
  char* ws = (char*)d_ws;
  size_t cur = 0;
  auto alloc = [&](size_t bytes)->char*{
    char* p = ws + cur;
    cur += (bytes + 255) & ~(size_t)255;
    return p;
  };
  _Float16* xf16 = (_Float16*)alloc((size_t)N0_ * FIN_ * 2);
  _Float16* hf16 = (_Float16*)alloc((size_t)N1_ * HC1_ * 2);
  _Float16* wt1[4]; for (int i = 0; i < 4; ++i) wt1[i] = (_Float16*)alloc((size_t)FIN_ * HC1_ * 2);
  _Float16* wt2[4]; for (int i = 0; i < 4; ++i) wt2[i] = (_Float16*)alloc((size_t)HC1_ * HC2_ * 2);
  float*    q1  = (float*)alloc((size_t)N1_ * HC1_ * 4);
  float*    k1  = (float*)alloc((size_t)N0_ * HC1_ * 4);
  float*    v1  = (float*)alloc((size_t)N0_ * HC1_ * 4);
  float*    s1  = (float*)alloc((size_t)N1_ * HC1_ * 4);
  float*    o1  = (float*)alloc((size_t)N1_ * HC1_ * 4);
  float*    al1 = (float*)alloc((size_t)E1_ * 2 * 4);
  unsigned* me1 = (unsigned*)alloc((size_t)N1_ * 2 * 4);
  float*    am1 = (float*)alloc((size_t)N1_ * 2 * 4);
  float*    dn1 = (float*)alloc((size_t)N1_ * 2 * 4);
  float*    q2  = (float*)alloc((size_t)B_  * HC2_ * 4);
  float*    k2  = (float*)alloc((size_t)N1_ * HC2_ * 4);
  float*    v2  = (float*)alloc((size_t)N1_ * HC2_ * 4);
  float*    s2  = (float*)alloc((size_t)B_  * HC2_ * 4);
  float*    o2  = (float*)alloc((size_t)B_  * HC2_ * 4);
  float*    al2 = (float*)alloc((size_t)E2_ * 4);
  unsigned* me2 = (unsigned*)alloc((size_t)B_ * 4);
  float*    am2 = (float*)alloc((size_t)B_ * 4);
  float*    dn2 = (float*)alloc((size_t)B_ * 4);

  const unsigned ENC_NEG_INF = 0x007FFFFFu;   // fenc(-inf)

  auto gemm = [&](const _Float16* A, const _Float16* Bt, const float* bias, float* C,
                  int M, int N, int K){
    int grid = cdiv(M / 32, 8) * (N / 64);
    size_t shmem = (size_t)N * K * 2;   // full weight matrix in LDS
    k_gemm_wmma<<<grid, 256, shmem, stream>>>(A, Bt, bias, C, M, N, K);
  };

  for (int r = 0; r < R_; ++r){
    // ---- gather + cast features ----
    k_gather<<<cdiv(N0_ * FIN_, 256), 256, 0, stream>>>(features, n_ids + (size_t)r * N0_,
                                                        xf16, N0_);
    // ---- layer 1 weights -> f16 transposed ----
    const float* w1p[4] = {wq1, wk1, wv1, ws1};
    for (int i = 0; i < 4; ++i)
      k_castT<<<cdiv(FIN_ * HC1_, 256), 256, 0, stream>>>(w1p[i] + (size_t)r * FIN_ * HC1_,
                                                          wt1[i], FIN_, HC1_);
    // ---- layer 1 GEMMs (WMMA) ----
    gemm(xf16, wt1[0], bq1 + (size_t)r * HC1_, q1, N1_, HC1_, FIN_);
    gemm(xf16, wt1[1], bk1 + (size_t)r * HC1_, k1, N0_, HC1_, FIN_);
    gemm(xf16, wt1[2], bv1 + (size_t)r * HC1_, v1, N0_, HC1_, FIN_);
    gemm(xf16, wt1[3], bs1 + (size_t)r * HC1_, s1, N1_, HC1_, FIN_);
    // ---- layer 1 edge softmax ----
    k_fill<<<256, 256, 0, stream>>>(me1, ENC_NEG_INF, N1_ * 2);
    k_fill<<<256, 256, 0, stream>>>((unsigned*)dn1, 0u, N1_ * 2);
    k_fill<<<1024, 256, 0, stream>>>((unsigned*)o1, 0u, N1_ * HC1_);
    const int* src1p = ei1 + (size_t)r * 2 * E1_;
    const int* dst1p = src1p + E1_;
    k_edge_alpha<<<cdiv(E1_ * 2, 8), 256, 0, stream>>>(q1, k1, src1p, dst1p, al1, E1_, 2);
    k_edge_max<<<cdiv(E1_ * 2, 256), 256, 0, stream>>>(al1, dst1p, me1, E1_, 2);
    k_amax_dec<<<cdiv(N1_ * 2, 256), 256, 0, stream>>>(me1, am1, N1_ * 2);
    k_edge_exp<<<cdiv(E1_ * 2, 256), 256, 0, stream>>>(al1, dst1p, am1, dn1, E1_, 2);
    k_edge_accum<<<cdiv(E1_ * 2, 8), 256, 0, stream>>>(al1, src1p, dst1p, dn1, v1, o1, E1_, 2);
    // ---- beta gate + BN + ELU -> f16 activations ----
    k_beta_bn_elu<<<cdiv(N1_, 8), 256, 0, stream>>>(o1, s1, wbeta1 + (size_t)r * 3 * HC1_,
        bng + (size_t)r * HC1_, bnb + (size_t)r * HC1_,
        bnm + (size_t)r * HC1_, bnv + (size_t)r * HC1_, hf16, N1_, HC1_);
    // ---- layer 2 weights -> f16 transposed ----
    const float* w2p[4] = {wq2, wk2, wv2, ws2};
    for (int i = 0; i < 4; ++i)
      k_castT<<<cdiv(HC1_ * HC2_, 256), 256, 0, stream>>>(w2p[i] + (size_t)r * HC1_ * HC2_,
                                                          wt2[i], HC1_, HC2_);
    // ---- layer 2 GEMMs (WMMA) ----
    gemm(hf16, wt2[0], bq2 + (size_t)r * HC2_, q2, B_,  HC2_, HC1_);
    gemm(hf16, wt2[1], bk2 + (size_t)r * HC2_, k2, N1_, HC2_, HC1_);
    gemm(hf16, wt2[2], bv2 + (size_t)r * HC2_, v2, N1_, HC2_, HC1_);
    gemm(hf16, wt2[3], bs2 + (size_t)r * HC2_, s2, B_,  HC2_, HC1_);
    // ---- layer 2 edge softmax ----
    k_fill<<<64, 256, 0, stream>>>(me2, ENC_NEG_INF, B_);
    k_fill<<<64, 256, 0, stream>>>((unsigned*)dn2, 0u, B_);
    k_fill<<<512, 256, 0, stream>>>((unsigned*)o2, 0u, B_ * HC2_);
    const int* src2p = ei2 + (size_t)r * 2 * E2_;
    const int* dst2p = src2p + E2_;
    k_edge_alpha<<<cdiv(E2_, 8), 256, 0, stream>>>(q2, k2, src2p, dst2p, al2, E2_, 1);
    k_edge_max<<<cdiv(E2_, 256), 256, 0, stream>>>(al2, dst2p, me2, E2_, 1);
    k_amax_dec<<<cdiv(B_, 256), 256, 0, stream>>>(me2, am2, B_);
    k_edge_exp<<<cdiv(E2_, 256), 256, 0, stream>>>(al2, dst2p, am2, dn2, E2_, 1);
    k_edge_accum<<<cdiv(E2_, 8), 256, 0, stream>>>(al2, src2p, dst2p, dn2, v2, o2, E2_, 1);
    // ---- final gated combine -> d_out[b, r*128 + c] ----
    k_beta_out<<<cdiv(B_, 8), 256, 0, stream>>>(o2, s2, wbeta2 + (size_t)r * 3 * HC2_,
        dout + (size_t)r * HC2_, R_ * HC2_, B_, HC2_);
  }
}
